// AttentionBlock_2534030705227
// MI455X (gfx1250) — matrix operand order repaired and checked by reading, not compiled, not run on hardware
//
#include <hip/hip_runtime.h>
#include <math.h>
#include <stdint.h>

// ---------------- problem constants ----------------
static constexpr int kB   = 2;
static constexpr int kS   = 2048;
static constexpr int kHID = 2048;
static constexpr int kNH  = 16;
static constexpr int kHD  = 128;
static constexpr int kM   = kB * kS;          // 4096 rows
#define SOFT_SCALE 0.08838834764831845f        // 1/sqrt(128)
#define NEG_INF    (-1e30f)

// ---------------- types ----------------
typedef __bf16 bf16;
typedef __attribute__((ext_vector_type(16))) __bf16 v16bf;
typedef __attribute__((ext_vector_type(8)))  __bf16 bf16x8;
typedef __attribute__((ext_vector_type(8)))  float  v8f;
typedef uint32_t u32x4 __attribute__((ext_vector_type(4)));
typedef uint32_t u32x8 __attribute__((ext_vector_type(8)));

__device__ __forceinline__ v16bf concat8(bf16x8 lo, bf16x8 hi) {
  return __builtin_shufflevector(lo, hi, 0,1,2,3,4,5,6,7,8,9,10,11,12,13,14,15);
}

// TDM: async-load a 32x128 bf16 tile (row pitch kHID elements) into LDS.
// 2-group D#: group0 = {count/lds_addr/global_addr/type}, group1 = dims/strides.
__device__ __forceinline__ void tdm_load_k_tile(const bf16* gptr, uint32_t lds_addr) {
  const uint64_t ga = (uint64_t)(uintptr_t)gptr;
  u32x4 g0;
  g0[0] = 1u;                                   // count=1 (valid), no gather
  g0[1] = lds_addr;                             // LDS byte address
  g0[2] = (uint32_t)ga;                         // global_addr[31:0]
  g0[3] = ((uint32_t)(ga >> 32) & 0x01FFFFFFu)  // global_addr[56:32]
          | 0x80000000u;                        // type=2 ("image") at [127:126]
  u32x8 g1;
  g1[0] = 0x00010000u;       // workgroup_mask=0, data_size=1 (2 bytes)
  g1[1] = 128u << 16;        // tensor_dim0 = 128 (bits[79:48] lo16)
  g1[2] = 0x80000000u;       // tensor_dim1 = 0x8000 rows (bits[111:80] lo16)
  g1[3] = 128u << 16;        // tensor_dim1 hi=0, tile_dim0 = 128
  g1[4] = 32u;               // tile_dim1 = 32, tile_dim2 = 0
  g1[5] = (uint32_t)kHID;    // tensor_dim0_stride = 2048 (lo32)
  g1[6] = 0u;                // stride hi + tensor_dim1_stride lo
  g1[7] = 0u;
  asm volatile("tensor_load_to_lds %0, %1" :: "s"(g0), "s"(g1) : "memory");
}

// ---------------- fp32 -> bf16 convert ----------------
__global__ __launch_bounds__(256)
void f32_to_bf16(const float* __restrict__ src, bf16* __restrict__ dst, int n) {
  int i = blockIdx.x * blockDim.x + threadIdx.x;
  int stride = gridDim.x * blockDim.x;
  for (; i < n; i += stride) dst[i] = (bf16)src[i];
}

// ---------------- GEMM helpers: ping-pong fragment sets ----------------
struct FragSet {
  bf16x8 alo, ahi;
  bf16x8 blo[4], bhi[4];
};

__device__ __forceinline__ void load_set(FragSet& s, const bf16* arow,
                                         const bf16* w0, const bf16* w1,
                                         const bf16* w2, const bf16* w3, int kk) {
  s.alo    = *(const bf16x8*)(arow + kk);
  s.ahi    = *(const bf16x8*)(arow + kk + 8);
  s.blo[0] = *(const bf16x8*)(w0 + kk);
  s.bhi[0] = *(const bf16x8*)(w0 + kk + 8);
  s.blo[1] = *(const bf16x8*)(w1 + kk);
  s.bhi[1] = *(const bf16x8*)(w1 + kk + 8);
  s.blo[2] = *(const bf16x8*)(w2 + kk);
  s.bhi[2] = *(const bf16x8*)(w2 + kk + 8);
  s.blo[3] = *(const bf16x8*)(w3 + kk);
  s.bhi[3] = *(const bf16x8*)(w3 + kk + 8);
}

__device__ __forceinline__ void wmma_set(v8f acc[4], const FragSet& s) {
  v16bf va = concat8(s.alo, s.ahi);
#pragma unroll
  for (int f = 0; f < 4; ++f)
    acc[f] = __builtin_amdgcn_wmma_f32_16x16x32_bf16(
        false, va, false, concat8(s.blo[f], s.bhi[f]), (short)0, acc[f],
        false, false);
}

// ---------------- GEMM: C[M,N] (f32) = A[M,K] (bf16) * W[N,K]^T (bf16) ------
// Wave tile 16x64 (4 accumulators), block = 8 waves -> 32x256 tile.
// Ping-pong double-buffered k-loop: loads for slice k+32 are in flight while
// WMMAs consume slice k; no register rotation between sets.
__global__ __launch_bounds__(256)
void gemm_bf16_nt(const bf16* __restrict__ A, const bf16* __restrict__ W,
                  float* __restrict__ C, int Kdim, int Ndim) {
  const int tid  = threadIdx.x;
  const int lane = tid & 31;
  const int wave = tid >> 5;
  const int wm   = wave & 1;            // 2 waves along M
  const int wn   = wave >> 1;           // 4 waves along N
  const int m0   = blockIdx.y * 32 + wm * 16;
  const int n0   = blockIdx.x * 256 + wn * 64;
  const int cl   = lane & 15;
  const int half = lane >> 4;
  const int khA  = half * 16;           // A: K 0-15 vs 16-31
  const int khB  = half * 16;           // B: K 0-15 vs 16-31

  const bf16* arow = A + (size_t)(m0 + cl) * Kdim + khA;
  const bf16* w0   = W + (size_t)(n0 + cl) * Kdim + khB;
  const bf16* w1   = w0 + (size_t)16 * Kdim;
  const bf16* w2   = w0 + (size_t)32 * Kdim;
  const bf16* w3   = w0 + (size_t)48 * Kdim;

  v8f acc[4] = {};
  FragSet s0, s1;
  load_set(s0, arow, w0, w1, w2, w3, 0);

  int k0 = 0;
#pragma unroll 1
  for (; k0 + 64 < Kdim; k0 += 64) {
    load_set(s1, arow, w0, w1, w2, w3, k0 + 32);
    wmma_set(acc, s0);
    load_set(s0, arow, w0, w1, w2, w3, k0 + 64);
    wmma_set(acc, s1);
  }
  load_set(s1, arow, w0, w1, w2, w3, k0 + 32);
  wmma_set(acc, s0);
  wmma_set(acc, s1);

  const int rb = half * 8;
#pragma unroll
  for (int f = 0; f < 4; ++f)
#pragma unroll
    for (int r = 0; r < 8; ++r)
      C[(size_t)(m0 + rb + r) * Ndim + n0 + f * 16 + cl] = acc[f][r];
}

// ---------------- fused RoPE + per-head RMSNorm, f32 -> bf16 ----------------
__global__ __launch_bounds__(128)
void rope_rmsnorm(const float* __restrict__ src, const float* __restrict__ w,
                  bf16* __restrict__ dst) {
  const int h   = blockIdx.x & (kNH - 1);
  const int tok = blockIdx.x >> 4;            // b*S + s
  const int pos = tok & (kS - 1);
  const int d   = threadIdx.x;
  const size_t base = (size_t)tok * kHID + (size_t)h * kHD;

  const float xv = src[base + d];
  const int   pd = d & 63;
  // freq = exp(log(10000) * 2*pd)  (no /head_dim, per reference); 1/inf -> 0
  const float invf = 1.0f / expf(9.210340371976184f * (float)(2 * pd));
  const float ang  = (float)pos * invf;
  const float c = cosf(ang), sn = sinf(ang);
  const float partner = src[base + ((d < 64) ? d + 64 : d - 64)];
  const float y = (d < 64) ? (xv * c - partner * sn) : (xv * c + partner * sn);

  float v = y * y;
#pragma unroll
  for (int off = 16; off > 0; off >>= 1) v += __shfl_xor(v, off, 32);
  __shared__ float red[4];
  if ((threadIdx.x & 31) == 0) red[threadIdx.x >> 5] = v;
  __syncthreads();
  const float ms = (red[0] + red[1] + red[2] + red[3]) * (1.0f / 128.0f);
  dst[base + d] = (bf16)(y * rsqrtf(ms + 1e-6f) * w[d]);
}

// ---------------- flash attention ----------------
// Grid: (S/128, NH, B); 8 waves/block, each owns a 16-row q tile.
// K tiles: TDM async load, double-buffered. V tiles: register-prefetched,
// transposed into LDS, double-buffered. One barrier per 32-key chunk.
__global__ __launch_bounds__(256)
void attention_kernel(const bf16* __restrict__ Q, const bf16* __restrict__ K,
                      const bf16* __restrict__ V,
                      const unsigned char* __restrict__ qmask,
                      bf16* __restrict__ O) {
  __shared__ bf16 Kbuf[2][32 * 128];   // [key][d] row-major (TDM-filled)
  __shared__ bf16 Vt[2][128 * 32];     // [d][key] transposed
  __shared__ bf16 Pbuf[8][16 * 32];    // per-wave P staging

  const int tid  = threadIdx.x;
  const int lane = tid & 31;
  const int wave = tid >> 5;
  const int b    = blockIdx.z;
  const int h    = blockIdx.y;
  const int q0b  = blockIdx.x * 128;
  const int q0   = q0b + wave * 16;
  const int cl   = lane & 15;
  const int half = lane >> 4;
  const int khA  = half * 16, khB = half * 16, rb = half * 8;

  // Q fragments (16 rows x 128 dims -> 4 K-chunks of 32)
  const bf16* qrow = Q + (size_t)(b * kS + q0 + cl) * kHID + h * kHD;
  v16bf va[4];
#pragma unroll
  for (int c = 0; c < 4; ++c) {
    bf16x8 lo = *(const bf16x8*)(qrow + c * 32 + khA);
    bf16x8 hi = *(const bf16x8*)(qrow + c * 32 + khA + 8);
    va[c] = concat8(lo, hi);
  }

  v8f o[8] = {};
  float mrow[8], lrow[8];
#pragma unroll
  for (int r = 0; r < 8; ++r) { mrow[r] = NEG_INF; lrow[r] = 0.0f; }

  const int nch   = q0b / 32 + 4;      // uniform across block (causal bound)
  const int ldrow = tid >> 3;          // 0..31
  const int ldcol = (tid & 7) * 16;    // 0..112
  const bf16* kbase = K + (size_t)b * kS * kHID + h * kHD;
  const bf16* vbase = V + (size_t)(b * kS + ldrow) * kHID + h * kHD + ldcol;

  // ---- prologue: chunk 0 ----
  if (wave == 0) tdm_load_k_tile(kbase, (uint32_t)(uintptr_t)&Kbuf[0][0]);
  bf16x8 pv1 = *(const bf16x8*)(vbase);
  bf16x8 pv2 = *(const bf16x8*)(vbase + 8);
  {
#pragma unroll
    for (int i = 0; i < 8; ++i) {
      Vt[0][(ldcol + i) * 32 + ldrow]     = pv1[i];
      Vt[0][(ldcol + 8 + i) * 32 + ldrow] = pv2[i];
    }
  }
  if (wave == 0) __builtin_amdgcn_s_wait_tensorcnt(0);
  __syncthreads();

  for (int ck = 0; ck < nch; ++ck) {
    const int  k0   = ck * 32;
    const int  buf  = ck & 1;
    const int  nbuf = buf ^ 1;
    const bool more = (ck + 1 < nch);

    if (more) {  // prefetch chunk ck+1 while computing ck
      if (wave == 0)
        tdm_load_k_tile(kbase + (size_t)(k0 + 32) * kHID,
                        (uint32_t)(uintptr_t)&Kbuf[nbuf][0]);
      pv1 = *(const bf16x8*)(vbase + (size_t)(k0 + 32) * kHID);
      pv2 = *(const bf16x8*)(vbase + (size_t)(k0 + 32) * kHID + 8);
    }

    // S = Q * K^T  (16x32 = two 16x16 tiles, K-depth 128)
    v8f sf[2] = {};
#pragma unroll
    for (int j = 0; j < 2; ++j)
#pragma unroll
      for (int c = 0; c < 4; ++c) {
        bf16x8 lo = *(const bf16x8*)&Kbuf[buf][(j * 16 + cl) * 128 + c * 32 + khB];
        bf16x8 hi = *(const bf16x8*)&Kbuf[buf][(j * 16 + cl) * 128 + c * 32 + khB + 8];
        sf[j] = __builtin_amdgcn_wmma_f32_16x16x32_bf16(
            false, va[c], false, concat8(lo, hi), (short)0, sf[j], false, false);
      }

    // causal + key mask, online softmax update
    float p0[8], p1[8];
    const int  key0 = k0 + cl, key1 = k0 + 16 + cl;
    const bool km0  = qmask[b * kS + key0] != 0;
    const bool km1  = qmask[b * kS + key1] != 0;
#pragma unroll
    for (int r = 0; r < 8; ++r) {
      const int qr = q0 + rb + r;
      float s0 = (km0 && key0 <= qr) ? sf[0][r] * SOFT_SCALE : NEG_INF;
      float s1 = (km1 && key1 <= qr) ? sf[1][r] * SOFT_SCALE : NEG_INF;
      float rm = fmaxf(s0, s1);
      for (int off = 1; off < 16; off <<= 1) rm = fmaxf(rm, __shfl_xor(rm, off, 16));
      const float mn    = fmaxf(mrow[r], rm);
      const float alpha = __expf(mrow[r] - mn);
      const float e0 = __expf(s0 - mn), e1 = __expf(s1 - mn);
      float rs = e0 + e1;
      for (int off = 1; off < 16; off <<= 1) rs += __shfl_xor(rs, off, 16);
      lrow[r] = lrow[r] * alpha + rs;
      mrow[r] = mn;
      p0[r] = e0; p1[r] = e1;
#pragma unroll
      for (int f = 0; f < 8; ++f) o[f][r] *= alpha;
    }

    // stage P through per-wave LDS to get A-fragment layout
#pragma unroll
    for (int r = 0; r < 8; ++r) {
      Pbuf[wave][(rb + r) * 32 + cl]      = (bf16)p0[r];
      Pbuf[wave][(rb + r) * 32 + 16 + cl] = (bf16)p1[r];
    }
    asm volatile("s_wait_dscnt 0" ::: "memory");
    bf16x8 plo = *(const bf16x8*)&Pbuf[wave][cl * 32 + khA];
    bf16x8 phi = *(const bf16x8*)&Pbuf[wave][cl * 32 + khA + 8];
    v16bf  pa  = concat8(plo, phi);

    // O += P * V   (16x128, K-depth 32)
#pragma unroll
    for (int f = 0; f < 8; ++f) {
      bf16x8 vlo = *(const bf16x8*)&Vt[buf][(f * 16 + cl) * 32 + khB];
      bf16x8 vhi = *(const bf16x8*)&Vt[buf][(f * 16 + cl) * 32 + khB + 8];
      o[f] = __builtin_amdgcn_wmma_f32_16x16x32_bf16(
          false, pa, false, concat8(vlo, vhi), (short)0, o[f], false, false);
    }

    if (more) {  // publish prefetched chunk; one barrier per chunk
#pragma unroll
      for (int i = 0; i < 8; ++i) {
        Vt[nbuf][(ldcol + i) * 32 + ldrow]     = pv1[i];
        Vt[nbuf][(ldcol + 8 + i) * 32 + ldrow] = pv2[i];
      }
      if (wave == 0) __builtin_amdgcn_s_wait_tensorcnt(0);
      __syncthreads();
    }
  }

  // epilogue: 1/l, zero masked query rows, write bf16
#pragma unroll
  for (int r = 0; r < 8; ++r) {
    const int   qr = q0 + rb + r;
    const float sc = qmask[b * kS + qr] ? (1.0f / lrow[r]) : 0.0f;
#pragma unroll
    for (int f = 0; f < 8; ++f)
      O[(size_t)(b * kS + qr) * kHID + h * kHD + f * 16 + cl] = (bf16)(o[f][r] * sc);
  }
}

// ---------------- host launcher ----------------
extern "C" void kernel_launch(void* const* d_in, const int* in_sizes, int n_in,
                              void* d_out, int out_size, void* d_ws, size_t ws_size,
                              hipStream_t stream) {
  (void)in_sizes; (void)n_in; (void)out_size; (void)ws_size;
  const float* x   = (const float*)d_in[0];
  const float* w_q = (const float*)d_in[1];
  const float* w_k = (const float*)d_in[2];
  const float* w_v = (const float*)d_in[3];
  const float* w_o = (const float*)d_in[4];
  const float* qnw = (const float*)d_in[5];
  const float* knw = (const float*)d_in[6];
  const unsigned char* qmask = (const unsigned char*)d_in[7];
  float* out = (float*)d_out;

  char* p = (char*)d_ws;
  bf16*  xb  = (bf16*)p;  p += (size_t)kM * kHID * 2;     // 16 MB
  bf16*  wqb = (bf16*)p;  p += (size_t)kHID * kHID * 2;   //  8 MB
  bf16*  wkb = (bf16*)p;  p += (size_t)kHID * kHID * 2;
  bf16*  wvb = (bf16*)p;  p += (size_t)kHID * kHID * 2;
  bf16*  wob = (bf16*)p;  p += (size_t)kHID * kHID * 2;
  float* Qf  = (float*)p; p += (size_t)kM * kHID * 4;     // 32 MB
  float* Kf  = (float*)p; p += (size_t)kM * kHID * 4;
  float* Vf  = (float*)p; p += (size_t)kM * kHID * 4;
  // stream-ordered aliases (producer of alias runs after last consumer of host)
  bf16* Qb    = xb;            // xb dead after QKV GEMMs
  bf16* Kb    = (bf16*)Qf;     // Qf dead after rope_rmsnorm(Q)
  bf16* Vb    = (bf16*)Kf;     // Kf dead after rope_rmsnorm(K)
  bf16* attnb = (bf16*)Vf;     // Vf dead after V convert

  const int nX = kM * kHID, nW = kHID * kHID;
  f32_to_bf16<<<1024, 256, 0, stream>>>(x,   xb,  nX);
  f32_to_bf16<<<1024, 256, 0, stream>>>(w_q, wqb, nW);
  f32_to_bf16<<<1024, 256, 0, stream>>>(w_k, wkb, nW);
  f32_to_bf16<<<1024, 256, 0, stream>>>(w_v, wvb, nW);
  f32_to_bf16<<<1024, 256, 0, stream>>>(w_o, wob, nW);

  dim3 gg(kHID / 256, kM / 32);
  gemm_bf16_nt<<<gg, 256, 0, stream>>>(xb, wqb, Qf, kHID, kHID);
  gemm_bf16_nt<<<gg, 256, 0, stream>>>(xb, wkb, Kf, kHID, kHID);
  gemm_bf16_nt<<<gg, 256, 0, stream>>>(xb, wvb, Vf, kHID, kHID);

  rope_rmsnorm<<<kM * kNH, 128, 0, stream>>>(Qf, qnw, Qb);
  rope_rmsnorm<<<kM * kNH, 128, 0, stream>>>(Kf, knw, Kb);
  f32_to_bf16<<<1024, 256, 0, stream>>>(Vf, Vb, nX);

  dim3 ag(kS / 128, kNH, kB);
  attention_kernel<<<ag, 256, 0, stream>>>(Qb, Kb, Vb, qmask, attnb);

  gemm_bf16_nt<<<gg, 256, 0, stream>>>(attnb, wob, out, kHID, kHID);
}
